// ScoreModel_1692217115002
// MI455X (gfx1250) — matrix operand order repaired and checked by reading, not compiled, not run on hardware
//
#include <hip/hip_runtime.h>
#include <hip/hip_bf16.h>

// ---------------------------------------------------------------------------
// ScoreModel cross-attention scorer for MI455X (gfx1250, wave32, WMMA).
//   K1: t_part = txt_hidden(2048x1024) @ W1[:1024](1024x512)
//       - 16x64 strip per wave, W1 panel staged to LDS with
//         GLOBAL_LOAD_ASYNC_TO_LDS_B128 (double buffered, ASYNCcnt)
//   K2: v_part = img_hidden(64x256) @ W1[1024:](256x512)       [VALU]
//   K3: gate   = softmax_L( W2 . prelu(t+v+b1) + b2, mask )    [VALU + LDS]
//   K4: out    = sum_i gate * max_h ( E[m] @ R[n]^T )          [WMMA f16->f32]
// ---------------------------------------------------------------------------

typedef __attribute__((ext_vector_type(16))) _Float16 v16h;
typedef __attribute__((ext_vector_type(8)))  float    v8f;

static constexpr int Mtxt = 64;     // text items
static constexpr int Nimg = 64;     // image items
static constexpr int L    = 32;     // tokens per text
static constexpr int H    = 36;     // image regions (excl. CLS)
static constexpr int DT   = 1024;   // txt hidden dim
static constexpr int D    = 256;    // embedding dim
static constexpr int O1   = 512;    // MLP hidden
static constexpr float NEGV = -1e30f;

// ---- CDNA5 async copy: global -> LDS, 16 bytes per lane (ASYNCcnt tracked)
__device__ __forceinline__ void async_g2l_b128(void* lds_dst, const void* gsrc) {
  // Generic pointer to a __shared__ object keeps the LDS offset in its low
  // 32 bits (ISA 10.2 aperture mapping) -- that is what VDST expects.
  const unsigned int loff = (unsigned int)(unsigned long long)lds_dst;
  asm volatile("global_load_async_to_lds_b128 %0, %1, off"
               :: "v"(loff), "v"(gsrc)
               : "memory");
}
__device__ __forceinline__ void wait_async0() {
  asm volatile("s_wait_asynccnt 0x0" ::: "memory");
}

// ---- WMMA fragment loaders (ISA 7.12.2 layout)

// A fragment: 16x32 (MxK) f16 from f32 row-major. Lane l holds row M = l&15.
__device__ __forceinline__ v16h load_a_frag_f32(const float* __restrict__ src,
                                                int ld, int row0, int k0, int lane) {
  const int m    = lane & 15;
  const int koff = (lane & 16) ? 8 : 0;
  const float* p = src + (size_t)(row0 + m) * (size_t)ld + (size_t)(k0 + koff);
  v16h a;
#pragma unroll
  for (int pr = 0; pr < 8; ++pr) {
    const int kb = (pr < 4) ? (pr * 2) : (16 + (pr - 4) * 2);
    a[2 * pr]     = (_Float16)p[kb];
    a[2 * pr + 1] = (_Float16)p[kb + 1];
  }
  return a;
}

// B fragment: 32x16 (KxN) f16 from an LDS panel sBk[32][64] (f32).
__device__ __forceinline__ v16h lds_b_frag(const float* sBk, int n0, int lane) {
  const int kb = (lane & 16) ? 8 : 0;
  const float* p = sBk + n0 + (lane & 15);
  v16h b;
#pragma unroll
  for (int j = 0; j < 16; ++j) b[j] = (_Float16)p[(kb + j + ((j >= 8) ? 8 : 0)) * 64];
  return b;
}

// B fragment from transposed f32 memory Bt[n][k], column clamp (n>=nmax -> 0).
__device__ __forceinline__ v16h load_bt_frag_clamp(const float* __restrict__ bt,
                                                   int ld, int k0, int n0, int nmax,
                                                   int lane) {
  const int kb = (lane & 16) ? 8 : 0;
  const int n = n0 + (lane & 15);
  v16h b;
#pragma unroll
  for (int j = 0; j < 16; ++j) {
    const int k = kb + j + ((j >= 8) ? 8 : 0);
    const float v = (n < nmax) ? bt[(size_t)n * (size_t)ld + (size_t)(k0 + k)] : 0.0f;
    b[j] = (_Float16)v;
  }
  return b;
}

__device__ __forceinline__ v8f wmma_f16(v16h a, v16h b, v8f c) {
  return __builtin_amdgcn_wmma_f32_16x16x32_f16(false, a, false, b,
                                                (short)0, c, false, false);
}

// ---------------------------------------------------------------------------
// K1: t_part[2048][512] = txt_hidden[2048][1024] @ W1[:1024][512]
// Block = 8 waves. Block tile: 128 rows x 64 cols. Wave tile: 16 x 64.
// W1 k-panel (32x64 f32, 8 KB) async-staged to LDS, double buffered.
// Grid: 16 M-groups x 8 N-groups = 128 blocks.
// ---------------------------------------------------------------------------
__global__ void __launch_bounds__(256)
k_tpart_gemm(const float* __restrict__ txt_hidden,   // 2048 x 1024
             const float* __restrict__ W1,           // 1280 x 512 (rows 0..1023)
             float* __restrict__ t_part) {           // 2048 x 512
  __shared__ float sB[2][32 * 64];                   // 2 x 8 KB

  const int tid  = threadIdx.x;
  const int lane = tid & 31;
  const int wave = tid >> 5;
  const int mg   = blockIdx.x >> 3;                  // 0..15
  const int ng   = blockIdx.x & 7;                   // 0..7
  const int tileM  = mg * 8 + wave;                  // 0..127
  const int row0   = tileM * 16;
  const int n_base = ng * 64;

  // stage W1[k0..k0+31][n_base..n_base+63] into sB[buf] (8 floats / thread)
  const int sr = tid >> 3;                 // 0..31  (k row within panel)
  const int sc = (tid & 7) * 8;            // 0,8,...,56
  const float* gpan = W1 + (size_t)sr * O1 + n_base + sc;
  float* lpan = &sB[0][sr * 64 + sc];

  // prologue: stage k-panel 0 into buffer 0
  async_g2l_b128(lpan, gpan);
  async_g2l_b128(lpan + 4, gpan + 4);

  v8f acc[4];
#pragma unroll
  for (int t = 0; t < 4; ++t) acc[t] = (v8f){};

  const int NSTEP = DT / 32;               // 32 k-steps
  for (int kk = 0; kk < NSTEP; ++kk) {
    const int cur = kk & 1;
    wait_async0();
    __syncthreads();                       // panel[cur] ready; prev reads done

    if (kk + 1 < NSTEP) {                  // stage next panel into other buffer
      const float* gn = gpan + (size_t)(kk + 1) * 32 * O1;
      float* ln = &sB[cur ^ 1][sr * 64 + sc];
      async_g2l_b128(ln, gn);
      async_g2l_b128(ln + 4, gn + 4);
    }

    const v16h a = load_a_frag_f32(txt_hidden, DT, row0, kk * 32, lane);
#pragma unroll
    for (int t = 0; t < 4; ++t) {
      const v16h b = lds_b_frag(&sB[cur][0], t * 16, lane);
      acc[t] = wmma_f16(a, b, acc[t]);
    }
  }

  const int mrow0 = row0 + ((lane & 16) ? 8 : 0);
#pragma unroll
  for (int t = 0; t < 4; ++t) {
    const int col = n_base + t * 16 + (lane & 15);
#pragma unroll
    for (int r = 0; r < 8; ++r)
      t_part[(size_t)(mrow0 + r) * O1 + col] = acc[t][r];
  }
}

// ---------------------------------------------------------------------------
// K2: v_part[64][512] = img_embedding[:,0,:](64x256) @ W1[1024:](256x512)
// ---------------------------------------------------------------------------
__global__ void __launch_bounds__(256)
k_vpart(const float* __restrict__ img_embedding,     // 64 x 37 x 256
        const float* __restrict__ W1,                // 1280 x 512
        float* __restrict__ v_part) {                // 64 x 512
  const int idx = blockIdx.x * blockDim.x + threadIdx.x;
  if (idx >= Nimg * O1) return;
  const int n = idx / O1;
  const int o = idx % O1;
  const float* hid = img_embedding + (size_t)n * (H + 1) * D;   // CLS row
  const float* w   = W1 + (size_t)DT * O1;                      // rows 1024..1279
  float acc = 0.0f;
#pragma unroll 8
  for (int e = 0; e < D; ++e)
    acc = fmaf(hid[e], w[(size_t)e * O1 + o], acc);
  v_part[idx] = acc;
}

// ---------------------------------------------------------------------------
// K3: gate[m][n][i] = softmax_i( mask(W2 . prelu(t+v+b1) + b2) )
// One block per (m,n). 8 waves; wave w handles rows i = 4w..4w+3.
// ---------------------------------------------------------------------------
__global__ void __launch_bounds__(256)
k_gate(const float* __restrict__ t_part,    // 2048 x 512
       const float* __restrict__ v_part,    // 64 x 512
       const float* __restrict__ b1,        // 512
       const float* __restrict__ W2,        // 512
       const float* __restrict__ b2,        // 1
       const float* __restrict__ prelu_a,   // 1
       const int*   __restrict__ txt_len,   // 64
       float* __restrict__ gate) {          // 64 x 64 x 32
  __shared__ float s_s[L];
  const int m = blockIdx.x;
  const int n = blockIdx.y;
  const int lane = threadIdx.x & 31;
  const int wave = threadIdx.x >> 5;
  const float pa = prelu_a[0];

  const float* vp = v_part + (size_t)n * O1;

#pragma unroll
  for (int rr = 0; rr < 4; ++rr) {
    const int i = wave * 4 + rr;
    const float* tp = t_part + (size_t)(m * L + i) * O1;
    float acc = 0.0f;
#pragma unroll
    for (int j = 0; j < 16; ++j) {
      const int c = lane * 16 + j;
      const float z = tp[c] + vp[c] + b1[c];
      const float pz = (z >= 0.0f) ? z : pa * z;
      acc = fmaf(W2[c], pz, acc);
    }
#pragma unroll
    for (int off = 16; off > 0; off >>= 1)
      acc += __shfl_xor(acc, off, 32);
    if (lane == 0) s_s[i] = acc + b2[0];
  }
  __syncthreads();

  if (threadIdx.x < L) {            // wave 0 does the 32-wide softmax
    const int i = threadIdx.x;
    const int len = txt_len[m];
    float val = (i >= len) ? NEGV : s_s[i];
    float mx = val;
#pragma unroll
    for (int off = 16; off > 0; off >>= 1)
      mx = fmaxf(mx, __shfl_xor(mx, off, 32));
    float e = __expf(val - mx);
    float sum = e;
#pragma unroll
    for (int off = 16; off > 0; off >>= 1)
      sum += __shfl_xor(sum, off, 32);
    gate[((size_t)m * Nimg + n) * L + i] = e / sum;
  }
}

// ---------------------------------------------------------------------------
// K4: out[m][n] = sum_i gate[m][n][i] * max_h ( E[m](32x256) @ R[n]^T )
// One wave per (m,n). 2 M-tiles x 3 N-tiles (h padded 36->48, clamped).
// ---------------------------------------------------------------------------
__global__ void __launch_bounds__(32)
k_sim_out(const float* __restrict__ txt_embedding,   // 64 x 32 x 256
          const float* __restrict__ img_embedding,   // 64 x 37 x 256
          const float* __restrict__ gate,            // 64 x 64 x 32
          float* __restrict__ out) {                 // 64 x 64
  __shared__ float s_max[L];
  const int m = blockIdx.x;
  const int n = blockIdx.y;
  const int lane = threadIdx.x & 31;

  const float* E = txt_embedding + (size_t)m * L * D;           // 32 x 256
  const float* R = img_embedding + (size_t)n * (H + 1) * D + D; // 36 x 256

  v8f acc[2][3];
#pragma unroll
  for (int tm = 0; tm < 2; ++tm)
#pragma unroll
    for (int tn = 0; tn < 3; ++tn) acc[tm][tn] = (v8f){};

  for (int k0 = 0; k0 < D; k0 += 32) {
    v16h a0 = load_a_frag_f32(E, D, 0, k0, lane);
    v16h a1 = load_a_frag_f32(E, D, 16, k0, lane);
#pragma unroll
    for (int tn = 0; tn < 3; ++tn) {
      const v16h b = load_bt_frag_clamp(R, D, k0, tn * 16, H, lane);
      acc[0][tn] = wmma_f16(a0, b, acc[0][tn]);
      acc[1][tn] = wmma_f16(a1, b, acc[1][tn]);
    }
  }

  // max over h (N dimension), per output row i = tm*16 + r + (lane>=16 ? 8 : 0)
  const int hcol = lane & 15;
#pragma unroll
  for (int tm = 0; tm < 2; ++tm) {
#pragma unroll
    for (int r = 0; r < 8; ++r) {
      float mr = NEGV;
#pragma unroll
      for (int tn = 0; tn < 3; ++tn) {
        const int h = tn * 16 + hcol;
        if (h < H) mr = fmaxf(mr, acc[tm][tn][r]);
      }
#pragma unroll
      for (int off = 8; off > 0; off >>= 1)
        mr = fmaxf(mr, __shfl_xor(mr, off, 32));
      if (hcol == 0) {
        const int i = tm * 16 + r + ((lane & 16) ? 8 : 0);
        s_max[i] = mr;
      }
    }
  }
  __syncthreads();

  const float* g = gate + ((size_t)m * Nimg + n) * L;
  float t = s_max[lane] * g[lane];
#pragma unroll
  for (int off = 16; off > 0; off >>= 1)
    t += __shfl_xor(t, off, 32);
  if (lane == 0) out[(size_t)m * Nimg + n] = t;
}

// ---------------------------------------------------------------------------
// Host-side launcher
// ---------------------------------------------------------------------------
extern "C" void kernel_launch(void* const* d_in, const int* in_sizes, int n_in,
                              void* d_out, int out_size, void* d_ws, size_t ws_size,
                              hipStream_t stream) {
  const float* txt_embedding = (const float*)d_in[0]; // 64 x 32 x 256
  const float* txt_hidden    = (const float*)d_in[1]; // 64 x 32 x 1024
  const int*   txt_lengths   = (const int*)  d_in[2]; // 64
  const float* img_embedding = (const float*)d_in[3]; // 64 x 37 x 256
  const float* W1            = (const float*)d_in[4]; // 1280 x 512
  const float* b1            = (const float*)d_in[5]; // 512
  const float* prelu_a       = (const float*)d_in[6]; // 1
  const float* W2            = (const float*)d_in[7]; // 512
  const float* b2            = (const float*)d_in[8]; // 1
  float* out = (float*)d_out;                         // 64 x 64

  // workspace layout
  char* ws = (char*)d_ws;
  float* t_part = (float*)ws;                                   // 2048*512 f32
  float* v_part = (float*)(ws + (size_t)2048 * 512 * 4);        // 64*512 f32
  float* gate   = (float*)(ws + (size_t)2048 * 512 * 4
                              + (size_t)64 * 512 * 4);          // 64*64*32 f32

  // K1: 16 M-groups x 8 N-groups
  k_tpart_gemm<<<dim3(128), dim3(256), 0, stream>>>(txt_hidden, W1, t_part);

  // K2
  k_vpart<<<dim3((Nimg * O1 + 255) / 256), dim3(256), 0, stream>>>(
      img_embedding, W1, v_part);

  // K3
  k_gate<<<dim3(Mtxt, Nimg), dim3(256), 0, stream>>>(
      t_part, v_part, b1, W2, b2, prelu_a, txt_lengths, gate);

  // K4
  k_sim_out<<<dim3(Mtxt, Nimg), dim3(32), 0, stream>>>(
      txt_embedding, img_embedding, gate, out);
}
